// HierarchicalSoftmax_412316860848
// MI455X (gfx1250) — matrix operand order repaired and checked by reading, not compiled, not run on hardware
//
#include <hip/hip_runtime.h>
#include <hip/hip_bf16.h>

// ---------------------------------------------------------------------------
// Hierarchical softmax: logits = X @ concat(W0,W1,W2)^T  (4096x4368, K=2048)
// fp32-accurate via bf16 hi/lo split: acc += hi*hi + hi*lo + lo*hi using
// v_wmma_f32_16x16x32_bf16.  If d_ws is big enough (69.3 MB) the split is
// precomputed once (removes per-tile conversion VALU);
// otherwise falls back to the fused split-in-tile path.
// ---------------------------------------------------------------------------

typedef __attribute__((ext_vector_type(16))) __bf16 v16bf;
typedef __attribute__((ext_vector_type(8)))  __bf16 v8bf;
typedef __attribute__((ext_vector_type(4)))  __bf16 v4bf;
typedef __attribute__((ext_vector_type(8)))  float  v8f;

#define D_K   2048   // reduction dim
#define N_TOT 4368   // 16 + 256 + 4096 output neurons
#define M_TOT 4096   // batch rows
#define BM    128
#define BN    128
#define BK    32
#define LDT   40     // padded LDS row stride in bf16 elems (80B, 16B aligned)

// ---------------------------------------------------------------------------
// shared helpers
// ---------------------------------------------------------------------------

// split one float4 into bf16 hi + bf16 residual-lo, store into LDS rows
__device__ __forceinline__ void split_store(__bf16* __restrict__ hi,
                                            __bf16* __restrict__ lo,
                                            float4 v) {
    float vv[4] = {v.x, v.y, v.z, v.w};
#pragma unroll
    for (int j = 0; j < 4; ++j) {
        __bf16 h = (__bf16)vv[j];            // RNE round to bf16
        hi[j] = h;
        lo[j] = (__bf16)(vv[j] - (float)h);  // residual in bf16
    }
}

__device__ __forceinline__ v16bf cat8(v8bf a, v8bf b) {
    return __builtin_shufflevector(a, b, 0,1,2,3,4,5,6,7,8,9,10,11,12,13,14,15);
}

// fragment loads (WMMA VGPR layouts per CDNA5 ISA) + 24 WMMAs
__device__ __forceinline__ void frag_mma(const __bf16* sAhi, const __bf16* sAlo,
                                         const __bf16* sBhi, const __bf16* sBlo,
                                         int wm, int wn, int lrow, int lhalf,
                                         v8f (&acc)[2][4]) {
    v16bf aHi[2], aLo[2], bHi[4], bLo[4];
#pragma unroll
    for (int mi = 0; mi < 2; ++mi) {
        int base = (wm * 32 + mi * 16 + lrow) * LDT + lhalf * 8;
        aHi[mi] = cat8(*(const v8bf*)&sAhi[base], *(const v8bf*)&sAhi[base + 16]);
        aLo[mi] = cat8(*(const v8bf*)&sAlo[base], *(const v8bf*)&sAlo[base + 16]);
    }
#pragma unroll
    for (int ni = 0; ni < 4; ++ni) {
        int base = (wn * 64 + ni * 16 + lrow) * LDT + lhalf * 8;
        bHi[ni] = cat8(*(const v8bf*)&sBhi[base], *(const v8bf*)&sBhi[base + 16]);
        bLo[ni] = cat8(*(const v8bf*)&sBlo[base], *(const v8bf*)&sBlo[base + 16]);
    }
#pragma unroll
    for (int mi = 0; mi < 2; ++mi)
#pragma unroll
        for (int ni = 0; ni < 4; ++ni) {
            acc[mi][ni] = __builtin_amdgcn_wmma_f32_16x16x32_bf16(
                false, aHi[mi], false, bHi[ni], (short)0, acc[mi][ni], false, false);
            acc[mi][ni] = __builtin_amdgcn_wmma_f32_16x16x32_bf16(
                false, aHi[mi], false, bLo[ni], (short)0, acc[mi][ni], false, false);
            acc[mi][ni] = __builtin_amdgcn_wmma_f32_16x16x32_bf16(
                false, aLo[mi], false, bHi[ni], (short)0, acc[mi][ni], false, false);
        }
}

// store raw logits (C/D layout: VGPR r -> M = lhalf*8 + r, N = lane%16)
__device__ __forceinline__ void store_acc(float* __restrict__ out, int m0, int n0,
                                          int wm, int wn, int lrow, int lhalf,
                                          v8f (&acc)[2][4]) {
#pragma unroll
    for (int mi = 0; mi < 2; ++mi)
#pragma unroll
        for (int ni = 0; ni < 4; ++ni) {
            int gcol = n0 + wn * 64 + ni * 16 + lrow;
            if (gcol < N_TOT) {
                int grow = m0 + wm * 32 + mi * 16 + lhalf * 8;
#pragma unroll
                for (int r = 0; r < 8; ++r)
                    out[(size_t)(grow + r) * N_TOT + gcol] = acc[mi][ni][r];
            }
        }
}

// ---------------------------------------------------------------------------
// one-time split kernels: f32 -> (bf16 hi, bf16 residual lo) planes in d_ws
// ---------------------------------------------------------------------------
__global__ __launch_bounds__(256)
void hs_split_x(const float* __restrict__ X,
                __bf16* __restrict__ hi, __bf16* __restrict__ lo) {
    size_t i = (size_t)blockIdx.x * 256 + threadIdx.x;   // float4 index
    float4 v = ((const float4*)X)[i];
    float vv[4] = {v.x, v.y, v.z, v.w};
    v4bf h, l;
#pragma unroll
    for (int j = 0; j < 4; ++j) {
        h[j] = (__bf16)vv[j];
        l[j] = (__bf16)(vv[j] - (float)h[j]);
    }
    ((v4bf*)hi)[i] = h;
    ((v4bf*)lo)[i] = l;
}

__global__ __launch_bounds__(256)
void hs_split_w(const float* __restrict__ W0, const float* __restrict__ W1,
                const float* __restrict__ W2,
                __bf16* __restrict__ hi, __bf16* __restrict__ lo) {
    size_t i = (size_t)blockIdx.x * 256 + threadIdx.x;   // float4 idx, [4368][512]
    int n  = (int)(i >> 9);
    int c4 = (int)(i & 511) << 2;
    const float* src = (n < 16)  ? (W0 + (size_t)n * D_K)
                     : (n < 272) ? (W1 + (size_t)(n - 16) * D_K)
                                 : (W2 + (size_t)(n - 272) * D_K);
    float4 v = *(const float4*)(src + c4);
    float vv[4] = {v.x, v.y, v.z, v.w};
    v4bf h, l;
#pragma unroll
    for (int j = 0; j < 4; ++j) {
        h[j] = (__bf16)vv[j];
        l[j] = (__bf16)(vv[j] - (float)h[j]);
    }
    ((v4bf*)hi)[i] = h;
    ((v4bf*)lo)[i] = l;
}

// ---------------------------------------------------------------------------
// GEMM from precomputed bf16 planes: register-prefetch pipelined K loop
// ---------------------------------------------------------------------------
__global__ __launch_bounds__(256)
void hs_gemm_wmma_pre(const __bf16* __restrict__ Xhi, const __bf16* __restrict__ Xlo,
                      const __bf16* __restrict__ Whi, const __bf16* __restrict__ Wlo,
                      float* __restrict__ out) {
    __shared__ alignas(32) __bf16 sAhi[BM * LDT];
    __shared__ alignas(32) __bf16 sAlo[BM * LDT];
    __shared__ alignas(32) __bf16 sBhi[BN * LDT];
    __shared__ alignas(32) __bf16 sBlo[BN * LDT];

    const int t     = threadIdx.x;
    const int m0    = blockIdx.y * BM;
    const int n0    = blockIdx.x * BN;
    const int wave  = t >> 5;
    const int lane  = t & 31;
    const int lrow  = lane & 15;
    const int lhalf = lane >> 4;
    const int wm    = wave & 3;
    const int wn    = wave >> 2;

    // staging: per array, thread owns two 8-elem (16B) chunks
    const int row0 = t >> 2;                 // rows 0..63
    const int row1 = (t + 256) >> 2;         // rows 64..127
    const int c8   = (t & 3) << 3;           // chunk column in elems
    const int nc0  = min(n0 + row0, N_TOT - 1);
    const int nc1  = min(n0 + row1, N_TOT - 1);

    const __bf16* pAh0 = Xhi + (size_t)(m0 + row0) * D_K + c8;
    const __bf16* pAh1 = Xhi + (size_t)(m0 + row1) * D_K + c8;
    const __bf16* pAl0 = Xlo + (size_t)(m0 + row0) * D_K + c8;
    const __bf16* pAl1 = Xlo + (size_t)(m0 + row1) * D_K + c8;
    const __bf16* pBh0 = Whi + (size_t)nc0 * D_K + c8;
    const __bf16* pBh1 = Whi + (size_t)nc1 * D_K + c8;
    const __bf16* pBl0 = Wlo + (size_t)nc0 * D_K + c8;
    const __bf16* pBl1 = Wlo + (size_t)nc1 * D_K + c8;

    v8f acc[2][4];
#pragma unroll
    for (int mi = 0; mi < 2; ++mi)
#pragma unroll
        for (int ni = 0; ni < 4; ++ni)
            acc[mi][ni] = (v8f){0.f, 0.f, 0.f, 0.f, 0.f, 0.f, 0.f, 0.f};

    v8bf rAh0, rAh1, rAl0, rAl1, rBh0, rBh1, rBl0, rBl1;
    // prologue: global loads for chunk 0
    rAh0 = *(const v8bf*)(pAh0); rAh1 = *(const v8bf*)(pAh1);
    rAl0 = *(const v8bf*)(pAl0); rAl1 = *(const v8bf*)(pAl1);
    rBh0 = *(const v8bf*)(pBh0); rBh1 = *(const v8bf*)(pBh1);
    rBl0 = *(const v8bf*)(pBl0); rBl1 = *(const v8bf*)(pBl1);

    for (int kc = 0; kc < D_K; kc += BK) {
        __syncthreads();   // previous compute finished; LDS reusable
        *(v8bf*)&sAhi[row0 * LDT + c8] = rAh0;
        *(v8bf*)&sAhi[row1 * LDT + c8] = rAh1;
        *(v8bf*)&sAlo[row0 * LDT + c8] = rAl0;
        *(v8bf*)&sAlo[row1 * LDT + c8] = rAl1;
        *(v8bf*)&sBhi[row0 * LDT + c8] = rBh0;
        *(v8bf*)&sBhi[row1 * LDT + c8] = rBh1;
        *(v8bf*)&sBlo[row0 * LDT + c8] = rBl0;
        *(v8bf*)&sBlo[row1 * LDT + c8] = rBl1;
        if (kc + BK < D_K) {
            int k2 = kc + BK;  // prefetch next chunk; lands during WMMA phase
            rAh0 = *(const v8bf*)(pAh0 + k2); rAh1 = *(const v8bf*)(pAh1 + k2);
            rAl0 = *(const v8bf*)(pAl0 + k2); rAl1 = *(const v8bf*)(pAl1 + k2);
            rBh0 = *(const v8bf*)(pBh0 + k2); rBh1 = *(const v8bf*)(pBh1 + k2);
            rBl0 = *(const v8bf*)(pBl0 + k2); rBl1 = *(const v8bf*)(pBl1 + k2);
        }
        __syncthreads();   // LDS chunk visible
        frag_mma(sAhi, sAlo, sBhi, sBlo, wm, wn, lrow, lhalf, acc);
    }
    store_acc(out, m0, n0, wm, wn, lrow, lhalf, acc);
}

// ---------------------------------------------------------------------------
// fallback: fused split-in-tile GEMM (used when d_ws is too small)
// ---------------------------------------------------------------------------
__global__ __launch_bounds__(256)
void hs_gemm_wmma(const float* __restrict__ X,
                  const float* __restrict__ W0,
                  const float* __restrict__ W1,
                  const float* __restrict__ W2,
                  float* __restrict__ out) {
    __shared__ alignas(32) __bf16 sAhi[BM * LDT];
    __shared__ alignas(32) __bf16 sAlo[BM * LDT];
    __shared__ alignas(32) __bf16 sBhi[BN * LDT];
    __shared__ alignas(32) __bf16 sBlo[BN * LDT];

    const int t     = threadIdx.x;
    const int m0    = blockIdx.y * BM;
    const int n0    = blockIdx.x * BN;
    const int wave  = t >> 5;
    const int lane  = t & 31;
    const int lrow  = lane & 15;
    const int lhalf = lane >> 4;
    const int wm    = wave & 3;
    const int wn    = wave >> 2;

    v8f acc[2][4];
#pragma unroll
    for (int mi = 0; mi < 2; ++mi)
#pragma unroll
        for (int ni = 0; ni < 4; ++ni)
            acc[mi][ni] = (v8f){0.f, 0.f, 0.f, 0.f, 0.f, 0.f, 0.f, 0.f};

    for (int kc = 0; kc < D_K; kc += BK) {
#pragma unroll
        for (int i = 0; i < 4; ++i) {
            int f   = t + i * 256;
            int row = f >> 3;
            int c4  = (f & 7) << 2;
            float4 v = *(const float4*)(X + (size_t)(m0 + row) * D_K + kc + c4);
            split_store(&sAhi[row * LDT + c4], &sAlo[row * LDT + c4], v);
        }
#pragma unroll
        for (int i = 0; i < 4; ++i) {
            int f   = t + i * 256;
            int row = f >> 3;
            int c4  = (f & 7) << 2;
            int n   = min(n0 + row, N_TOT - 1);
            const float* wr = (n < 16)  ? (W0 + (size_t)n * D_K)
                            : (n < 272) ? (W1 + (size_t)(n - 16) * D_K)
                                        : (W2 + (size_t)(n - 272) * D_K);
            float4 v = *(const float4*)(wr + kc + c4);
            split_store(&sBhi[row * LDT + c4], &sBlo[row * LDT + c4], v);
        }
        __syncthreads();
        frag_mma(sAhi, sAlo, sBhi, sBlo, wm, wn, lrow, lhalf, acc);
        __syncthreads();
    }
    store_acc(out, m0, n0, wm, wn, lrow, lhalf, acc);
}

// ---------------------------------------------------------------------------
// Epilogue: one workgroup per batch row. Adds bias, per-group(16) log-softmax,
// chains parent log-probs, writes [lp0|lp1|lp2] in place and lp2 to region 2.
// ---------------------------------------------------------------------------
__global__ __launch_bounds__(256)
void hs_softmax_epilogue(const float* __restrict__ b0,
                         const float* __restrict__ b1,
                         const float* __restrict__ b2,
                         float* __restrict__ out,    // [M_TOT, N_TOT] in-place
                         float* __restrict__ out2)   // [M_TOT, 4096]
{
    __shared__ float s[N_TOT];
    const int t = threadIdx.x;
    const int b = blockIdx.x;
    float* row = out + (size_t)b * N_TOT;

    for (int i = t; i < N_TOT; i += 256) {
        float bias = (i < 16)  ? b0[i]
                   : (i < 272) ? b1[i - 16]
                               : b2[i - 272];
        s[i] = row[i] + bias;
    }
    __syncthreads();

    for (int g = t; g < 273; g += 256) {
        int base = g * 16;
        float m = s[base];
#pragma unroll
        for (int j = 1; j < 16; ++j) m = fmaxf(m, s[base + j]);
        float sum = 0.f;
#pragma unroll
        for (int j = 0; j < 16; ++j) sum += expf(s[base + j] - m);
        float lse = m + logf(sum);
#pragma unroll
        for (int j = 0; j < 16; ++j) s[base + j] -= lse;
    }
    __syncthreads();

    for (int i = t; i < 256; i += 256) s[16 + i] += s[i >> 4];
    __syncthreads();

    for (int i = t; i < 4096; i += 256) s[272 + i] += s[16 + (i >> 4)];
    __syncthreads();

    for (int i = t; i < N_TOT; i += 256) row[i] = s[i];
    float* r2 = out2 + (size_t)b * 4096;
    for (int i = t; i < 4096; i += 256) r2[i] = s[272 + i];
}

extern "C" void kernel_launch(void* const* d_in, const int* in_sizes, int n_in,
                              void* d_out, int out_size, void* d_ws, size_t ws_size,
                              hipStream_t stream) {
    (void)in_sizes; (void)n_in; (void)out_size;
    const float* x  = (const float*)d_in[0];
    const float* W0 = (const float*)d_in[1];
    const float* b0 = (const float*)d_in[2];
    const float* W1 = (const float*)d_in[3];
    const float* b1 = (const float*)d_in[4];
    const float* W2 = (const float*)d_in[5];
    const float* b2 = (const float*)d_in[6];

    float* out  = (float*)d_out;                       // [4096, 4368]
    float* out2 = out + (size_t)M_TOT * N_TOT;         // [4096, 4096]

    dim3 gGemm((N_TOT + BN - 1) / BN, M_TOT / BM);     // 35 x 32 workgroups

    const size_t XHL = (size_t)M_TOT * D_K;            // elems per X plane
    const size_t WHL = (size_t)N_TOT * D_K;            // elems per W plane
    const size_t need = (2 * XHL + 2 * WHL) * 2;       // bytes (bf16)

    if (ws_size >= need) {
        char* w = (char*)d_ws;
        __bf16* Xhi = (__bf16*)(w);
        __bf16* Xlo = (__bf16*)(w + XHL * 2);
        __bf16* Whi = (__bf16*)(w + 2 * XHL * 2);
        __bf16* Wlo = (__bf16*)(w + (2 * XHL + WHL) * 2);
        hs_split_x<<<(unsigned)(XHL / 4 / 256), 256, 0, stream>>>(x, Xhi, Xlo);
        hs_split_w<<<(unsigned)(WHL / 4 / 256), 256, 0, stream>>>(W0, W1, W2, Whi, Wlo);
        hs_gemm_wmma_pre<<<gGemm, 256, 0, stream>>>(Xhi, Xlo, Whi, Wlo, out);
    } else {
        hs_gemm_wmma<<<gGemm, 256, 0, stream>>>(x, W0, W1, W2, out);
    }
    hs_softmax_epilogue<<<M_TOT, 256, 0, stream>>>(b0, b1, b2, out, out2);
}
